// GCN_57586921505017
// MI455X (gfx1250) — matrix operand order repaired and checked by reading, not compiled, not run on hardware
//
#include <hip/hip_runtime.h>
#include <hip/hip_bf16.h>
#include <stdint.h>

// ---------------------------------------------------------------------------
// GCN inference pipeline for gfx1250 (MI455X), wave32.
//   h = relu( Din^-1/2 * A * (Dout^-1/2 * x) * W1 + b1 )   (x2, then W3 layer)
//   out = log_softmax(...)
// GEMMs: v_wmma_f32_16x16x32_bf16 (f32 accumulate); A-tiles staged to LDS via
// GLOBAL_LOAD_ASYNC_TO_LDS_B128 (ASYNCcnt); weights pre-packed into per-lane
// bf16 B-fragments (branch-free 32B loads).
// ---------------------------------------------------------------------------

typedef __bf16 bf16_t;
typedef __attribute__((ext_vector_type(16))) __bf16 v16bf;
typedef __attribute__((ext_vector_type(8)))  float  v8f;

#define DHID 256
#define KTILES (DHID / 32)   // 8 WMMA K-steps of 32

// ---- zero a float buffer --------------------------------------------------
__global__ void gcn_zero_kernel(float* __restrict__ p, long long n) {
    long long i = (long long)blockIdx.x * blockDim.x + threadIdx.x;
    if (i < n) p[i] = 0.0f;
}

// ---- degree histogram (float atomics, matches segment_sum of ones) --------
__global__ void gcn_degree_kernel(const int* __restrict__ src,
                                  const int* __restrict__ dst,
                                  float* __restrict__ deg_out,
                                  float* __restrict__ deg_in,
                                  int E) {
    int i = blockIdx.x * blockDim.x + threadIdx.x;
    if (i < E) {
        atomicAdd(&deg_out[src[i]], 1.0f);
        atomicAdd(&deg_in[dst[i]], 1.0f);
    }
}

// ---- deg -> inv_sqrt in place ---------------------------------------------
__global__ void gcn_invsqrt_kernel(float* __restrict__ deg, long long n) {
    long long i = (long long)blockIdx.x * blockDim.x + threadIdx.x;
    if (i < n) {
        float d = deg[i];
        deg[i] = (d > 0.0f) ? rsqrtf(fmaxf(d, 1.0f)) : 0.0f;
    }
}

// ---- pack W [256 x Ncols] f32 into per-lane bf16 B-fragments --------------
// Fragment f = tn*KTILES + kc covers K rows [kc*32, kc*32+32), cols
// [tn*16, tn*16+16). Lane l holds 16 sequential-k values:
//   k = kc*32 + (l>>4)*16 + j,  col = tn*16 + (l&15),  j = 0..15
// stored contiguously at Wp[(f*32 + l)*16 + j]. Cols >= Ncols pack as 0.
__global__ void gcn_packB_kernel(const float* __restrict__ W,
                                 bf16_t* __restrict__ Wp,
                                 int Ncols, int nTiles) {
    int idx = blockIdx.x * blockDim.x + threadIdx.x;   // (frag, lane)
    int total = nTiles * KTILES * 32;
    if (idx >= total) return;
    int l  = idx & 31;
    int f  = idx >> 5;
    int kc = f % KTILES;
    int tn = f / KTILES;
    int col = tn * 16 + (l & 15);
    int kbase = kc * 32 + (l >> 4) * 16;
    bf16_t* o = Wp + (long long)idx * 16;
    #pragma unroll
    for (int j = 0; j < 16; ++j) {
        float w = (col < Ncols) ? W[(long long)(kbase + j) * Ncols + col] : 0.0f;
        o[j] = (bf16_t)w;
    }
}

// ---- fused SpMM scatter: agg[dst[e],:] += inv_out[src[e]] * in[src[e],:] --
// 64 threads per edge, each handles one float4 (256 floats per row).
__global__ void gcn_spmm_kernel(const float* __restrict__ in,
                                const float* __restrict__ inv_out,
                                const int* __restrict__ src,
                                const int* __restrict__ dst,
                                float* __restrict__ agg,
                                int E) {
    long long gtid = (long long)blockIdx.x * blockDim.x + threadIdx.x;
    long long e = gtid >> 6;
    int q = (int)(gtid & 63);
    if (e >= E) return;
    int s = src[e];
    int d = dst[e];
    float sc = inv_out[s];
    const float4 v = ((const float4*)in)[(long long)s * 64 + q];
    float* o = agg + (long long)d * DHID + q * 4;
    atomicAdd(o + 0, v.x * sc);
    atomicAdd(o + 1, v.y * sc);
    atomicAdd(o + 2, v.z * sc);
    atomicAdd(o + 3, v.w * sc);
}

// ---- WMMA GEMM: out[M,Ncols] = relu?( (ascale[m]*A[m,:]) @ W + bias ) -----
// A: [M x 256] f32; Wp: packed bf16 B-fragments. 16x256 f32 A-tile is staged
// into LDS once per block with async b128 loads, then 8 waves (one 16x16
// output tile each) run 8 fully-unrolled v_wmma_f32_16x16x32_bf16 over K=256.
__global__ __launch_bounds__(256)
void gcn_gemm_wmma_kernel(const float* __restrict__ A,
                          const float* __restrict__ ascale,
                          const bf16_t* __restrict__ Wp,
                          const float* __restrict__ bias,
                          float* __restrict__ out,
                          int M, int Ncols, int do_relu) {
    const int K = DHID;
    const int wave = threadIdx.x >> 5;
    const int lane = threadIdx.x & 31;
    const int nTiles = (Ncols + 15) >> 4;
    const int tm = blockIdx.x;

    const int half = lane >> 4;       // 0: lanes 0-15, 1: lanes 16-31
    const int mloc = lane & 15;

    // Clamped tile base: tile [tileBase, tileBase+16) always in range.
    int tileBase = tm * 16;
    if (tileBase + 16 > M) tileBase = (M > 16) ? (M - 16) : 0;

    // ---- stage contiguous 16KB A-tile into LDS (async, ASYNCcnt) ----
    __shared__ float smem[16 * DHID];                    // 16 KB
    {
        uint32_t sbase = (uint32_t)(uintptr_t)&smem[0];  // LDS byte offset
        const char* gbase = (const char*)(A + (long long)tileBase * K);
        #pragma unroll
        for (int i = 0; i < 4; ++i) {
            uint32_t boff = (uint32_t)(((wave * 4 + i) * 32 + lane) * 16);
            uint32_t ldsa = sbase + boff;
            const void* ga = gbase + boff;
            asm volatile("global_load_async_to_lds_b128 %0, %1, off"
                         :: "v"(ldsa), "v"(ga) : "memory");
        }
        asm volatile("s_wait_asynccnt 0x0" ::: "memory");
    }
    __syncthreads();

    const int tn = blockIdx.y * 8 + wave;
    if (tn >= nTiles) return;

    // A-fragment: lane half 0 holds K {0..15},
    // half 1 holds K {16..31}; M = mloc.
    int rowA = tileBase + mloc;
    float as = ascale[rowA];
    const float*  abase = smem + mloc * K + half * 16;   // LDS
    const bf16_t* bbase = Wp + ((long long)tn * KTILES * 32 + lane) * 16;

    v8f c = {};
    #pragma unroll
    for (int kc = 0; kc < KTILES; ++kc) {
        const float* arow = abase + kc * 32;
        float4 a0 = *(const float4*)(arow + 0);    // k+0..3
        float4 a1 = *(const float4*)(arow + 4);    // k+4..7
        float4 a2 = *(const float4*)(arow + 8);    // k+8..11
        float4 a3 = *(const float4*)(arow + 12);   // k+12..15
        v16bf b = *(const v16bf*)(bbase + (long long)kc * 32 * 16);

        v16bf a;
        a[0]  = (bf16_t)(a0.x * as);  a[1]  = (bf16_t)(a0.y * as);
        a[2]  = (bf16_t)(a0.z * as);  a[3]  = (bf16_t)(a0.w * as);
        a[4]  = (bf16_t)(a1.x * as);  a[5]  = (bf16_t)(a1.y * as);
        a[6]  = (bf16_t)(a1.z * as);  a[7]  = (bf16_t)(a1.w * as);
        a[8]  = (bf16_t)(a2.x * as);  a[9]  = (bf16_t)(a2.y * as);
        a[10] = (bf16_t)(a2.z * as);  a[11] = (bf16_t)(a2.w * as);
        a[12] = (bf16_t)(a3.x * as);  a[13] = (bf16_t)(a3.y * as);
        a[14] = (bf16_t)(a3.z * as);  a[15] = (bf16_t)(a3.w * as);

        c = __builtin_amdgcn_wmma_f32_16x16x32_bf16(
            /*neg_a=*/false, a, /*neg_b=*/false, b,
            /*c_mod=*/(short)0, c, /*reuse_a=*/false, /*reuse_b=*/false);
    }

    // C/D layout: VGPR r, lanes 0-15 -> M=r, N=lane; lanes 16-31 -> M=8+r.
    const int ng = tn * 16 + mloc;
    if (ng < Ncols) {
        float bv = bias[ng];
        #pragma unroll
        for (int r = 0; r < 8; ++r) {
            int mg = tileBase + half * 8 + r;
            float v = c[r] + bv;
            if (do_relu) v = fmaxf(v, 0.0f);
            out[(long long)mg * Ncols + ng] = v;
        }
    }
}

// ---- row-wise log_softmax, one wave32 per row (C <= 64) -------------------
__global__ void gcn_logsoftmax_kernel(float* __restrict__ out, int M, int C) {
    int wave = threadIdx.x >> 5;
    int lane = threadIdx.x & 31;
    int row = blockIdx.x * (blockDim.x >> 5) + wave;
    if (row >= M) return;
    float* p = out + (long long)row * C;
    const float NEG = -1e30f;
    float v0 = (lane < C) ? p[lane] : NEG;
    float v1 = (lane + 32 < C) ? p[lane + 32] : NEG;
    float m = fmaxf(v0, v1);
    #pragma unroll
    for (int o = 16; o > 0; o >>= 1) m = fmaxf(m, __shfl_xor(m, o, 32));
    float s = 0.0f;
    if (lane < C) s += __expf(v0 - m);
    if (lane + 32 < C) s += __expf(v1 - m);
    #pragma unroll
    for (int o = 16; o > 0; o >>= 1) s += __shfl_xor(s, o, 32);
    float lg = __logf(s);
    if (lane < C) p[lane] = v0 - m - lg;
    if (lane + 32 < C) p[lane + 32] = v1 - m - lg;
}

// ---------------------------------------------------------------------------
extern "C" void kernel_launch(void* const* d_in, const int* in_sizes, int n_in,
                              void* d_out, int out_size, void* d_ws, size_t ws_size,
                              hipStream_t stream) {
    const float* x  = (const float*)d_in[0];
    const float* W1 = (const float*)d_in[1];
    const float* b1 = (const float*)d_in[2];
    const float* W2 = (const float*)d_in[3];
    const float* b2 = (const float*)d_in[4];
    const float* W3 = (const float*)d_in[5];
    const float* b3 = (const float*)d_in[6];
    const int*  src = (const int*)d_in[7];
    const int*  dst = (const int*)d_in[8];

    const int N    = in_sizes[0] / DHID;   // 100000
    const int E    = in_sizes[7];          // 1600000
    const int DOUT = in_sizes[6];          // 40

    // workspace layout
    const int maxTiles = DHID / 16;                       // 16 col tiles
    bf16_t* Wp  = (bf16_t*)d_ws;                          // 16*8*32*16 bf16 = 128KB
    float*  deg = (float*)((char*)d_ws + (size_t)maxTiles * KTILES * 32 * 16 * sizeof(bf16_t));
    float*  inv_out = deg;                                // N
    float*  inv_in  = deg + N;                            // N
    float*  agg = deg + 2LL * N;                          // N*256
    float*  h   = agg + (long long)N * DHID;              // N*256

    const int T = 256;
    const long long ND = (long long)N * DHID;

    // ---- degrees -> inv-sqrt ----
    long long n2 = 2LL * N;
    gcn_zero_kernel<<<(unsigned)((n2 + T - 1) / T), T, 0, stream>>>(deg, n2);
    gcn_degree_kernel<<<(E + T - 1) / T, T, 0, stream>>>(src, dst, inv_out, inv_in, E);
    gcn_invsqrt_kernel<<<(unsigned)((n2 + T - 1) / T), T, 0, stream>>>(deg, n2);

    const unsigned gridND   = (unsigned)((ND + T - 1) / T);
    const unsigned gridSpmm = (unsigned)(((long long)E * 64 + T - 1) / T);
    const unsigned gridM    = (unsigned)((N + 15) / 16);
    const int packHid = maxTiles * KTILES * 32;           // threads to pack a 256-col W
    const int tilesOut = (DOUT + 15) / 16;
    const int packOut = tilesOut * KTILES * 32;

    // ---- layer 1: x -> h ----
    gcn_zero_kernel<<<gridND, T, 0, stream>>>(agg, ND);
    gcn_spmm_kernel<<<gridSpmm, T, 0, stream>>>(x, inv_out, src, dst, agg, E);
    gcn_packB_kernel<<<(packHid + T - 1) / T, T, 0, stream>>>(W1, Wp, DHID, maxTiles);
    gcn_gemm_wmma_kernel<<<dim3(gridM, 2), 256, 0, stream>>>(
        agg, inv_in, Wp, b1, h, N, DHID, 1);

    // ---- layer 2: h -> h ----
    gcn_zero_kernel<<<gridND, T, 0, stream>>>(agg, ND);
    gcn_spmm_kernel<<<gridSpmm, T, 0, stream>>>(h, inv_out, src, dst, agg, E);
    gcn_packB_kernel<<<(packHid + T - 1) / T, T, 0, stream>>>(W2, Wp, DHID, maxTiles);
    gcn_gemm_wmma_kernel<<<dim3(gridM, 2), 256, 0, stream>>>(
        agg, inv_in, Wp, b2, h, N, DHID, 1);

    // ---- layer 3: h -> d_out (DOUT=40 -> 3 col tiles, 1 y-block) ----
    gcn_zero_kernel<<<gridND, T, 0, stream>>>(agg, ND);
    gcn_spmm_kernel<<<gridSpmm, T, 0, stream>>>(h, inv_out, src, dst, agg, E);
    gcn_packB_kernel<<<(packOut + T - 1) / T, T, 0, stream>>>(W3, Wp, DOUT, tilesOut);
    gcn_gemm_wmma_kernel<<<dim3(gridM, 1), 256, 0, stream>>>(
        agg, inv_in, Wp, b3, (float*)d_out, N, DOUT, 0);

    // ---- log_softmax over rows of d_out ----
    gcn_logsoftmax_kernel<<<(N + 7) / 8, 256, 0, stream>>>((float*)d_out, N, DOUT);
}
